// RPN_29042568855566
// MI455X (gfx1250) — matrix operand order repaired and checked by reading, not compiled, not run on hardware
//
#include <hip/hip_runtime.h>
#include <hip/hip_bf16.h>
#include <math.h>

typedef __bf16 bf16;
typedef __attribute__((ext_vector_type(16))) __bf16 v16bf;
typedef __attribute__((ext_vector_type(8)))  __bf16 bf16x8;
typedef __attribute__((ext_vector_type(4)))  __bf16 bf16x4;
typedef __attribute__((ext_vector_type(8)))  float  v8f;

#define CIN    512
#define NPIX   16384
#define NANCH  147456
#define NT     64

union Frag { v16bf v; bf16x8 h[2]; };

// ---------------------------------------------------------------- converts
__global__ void k_cvt_x(const float* __restrict__ x, bf16* __restrict__ xb, int n) {
    int i = (blockIdx.x * blockDim.x + threadIdx.x) * 4;
    if (i < n) {
        float4 f = *(const float4*)(x + i);
        bf16x4 o;
        o[0] = (bf16)f.x; o[1] = (bf16)f.y; o[2] = (bf16)f.z; o[3] = (bf16)f.w;
        *(bf16x4*)(xb + i) = o;
    }
}

// conv_w (co,ci,3,3) fp32 -> wbf[co][tap][ci] bf16 ; combined 1x1 head weights wc[64][512]
__global__ void k_cvt_w(const float* __restrict__ cw, const float* __restrict__ bw,
                        const float* __restrict__ sw, bf16* __restrict__ wbf,
                        bf16* __restrict__ wc) {
    int i = blockIdx.x * blockDim.x + threadIdx.x;
    if (i < CIN * 9 * CIN) {
        int co = i / 4608; int rem = i - co * 4608;
        int tap = rem >> 9; int ci = rem & 511;
        wbf[i] = (bf16)cw[(size_t)co * 4608 + ci * 9 + tap];
    }
    if (i < 64 * CIN) {
        int m = i >> 9, ci = i & 511;
        float v = 0.f;
        if (m < 36)      v = bw[m * CIN + ci];
        else if (m < 54) v = sw[(m - 36) * CIN + ci];
        wc[i] = (bf16)v;
    }
}

// ------------------------------------------------- 3x3 conv as implicit GEMM
// WG tile: 128 cout x 128 pixels (one full image row). 8 waves, each 32x64.
// Double-buffered LDS pipeline: one barrier per 32-K chunk; global loads for
// chunk k+1 overlap the 8 WMMAs of chunk k. The 3x3 horizontal halo (+/-1) is
// applied by shifting the transposed LDS-store address, so staging stays in
// two bf16x8 VGPR vectors (no private arrays -> no scratch spills).
__global__ __launch_bounds__(256) void k_conv(const bf16* __restrict__ xb,
                                              const bf16* __restrict__ wbf,
                                              const float* __restrict__ conv_b,
                                              bf16* __restrict__ hbuf) {
    __shared__ __attribute__((aligned(16))) bf16 At[2][128][40];  // [co][k], pad 40
    __shared__ __attribute__((aligned(16))) bf16 Bt[2][128][40];  // [n][k] transposed

    int tid  = threadIdx.x;
    int nb   = blockIdx.x & 127, mb = blockIdx.x >> 7;
    int hrow = nb;
    int p0   = nb * 128;
    int co0  = mb * 128;

    int wv = tid >> 5, lane = tid & 31;
    int wm = wv & 3,  wn = wv >> 2;
    int mbase = wm * 32, nbase = wn * 64;
    int ml = lane & 15, sel = lane >> 4;

    v8f acc[2][4] = {};

    int am = tid >> 1, ah = (tid & 1) * 16;          // A tile: 128 x 32 halves
    int bk = tid >> 3, bn16 = (tid & 7) * 16;        // B tile: 32 x 128 halves

    // stage global tile for iteration `iter` into registers (vectors only)
    auto load_stage = [&](int iter, bf16x8& ra0, bf16x8& ra1,
                          bf16x8& rb0, bf16x8& rb1, bool& rowok) {
        int tap = iter >> 4;
        int ci0 = (iter & 15) << 5;
        int ky  = tap / 3;
        int hh  = hrow + ky - 1;
        const bf16* asrc = wbf + (size_t)(co0 + am) * 4608 + tap * 512 + ci0 + ah;
        ra0 = *(const bf16x8*)asrc;
        ra1 = *(const bf16x8*)(asrc + 8);
        rowok = (hh >= 0) && (hh < 128);
        if (rowok) {
            const bf16* rowp = xb + (size_t)(ci0 + bk) * NPIX + hh * 128;
            __builtin_prefetch(rowp + NPIX, 0, 1);
            rb0 = *(const bf16x8*)(rowp + bn16);
            rb1 = *(const bf16x8*)(rowp + bn16 + 8);
        } else {
            #pragma unroll
            for (int j = 0; j < 8; ++j) { rb0[j] = (bf16)0.f; rb1[j] = (bf16)0.f; }
        }
    };
    // commit staged registers into LDS buffer `buf`; horizontal halo shift in
    // the store address: element x[w] lands at n = w + (1 - kx)
    auto store_stage = [&](int buf, int iter, bf16x8 ra0, bf16x8 ra1,
                           bf16x8 rb0, bf16x8 rb1, bool rowok) {
        int tap = iter >> 4;
        int kx  = tap - (tap / 3) * 3;
        *(bf16x8*)&At[buf][am][ah]     = ra0;
        *(bf16x8*)&At[buf][am][ah + 8] = ra1;
        if (kx == 1 || !rowok) {                       // direct
            #pragma unroll
            for (int j = 0; j < 8; ++j) {
                Bt[buf][bn16 + j][bk]     = rb0[j];
                Bt[buf][bn16 + 8 + j][bk] = rb1[j];
            }
        } else if (kx == 0) {                          // shift right: n = w + 1
            #pragma unroll
            for (int j = 0; j < 8; ++j) Bt[buf][bn16 + j + 1][bk] = rb0[j];
            #pragma unroll
            for (int j = 0; j < 7; ++j) Bt[buf][bn16 + 9 + j][bk] = rb1[j];
            if (bn16 + 16 < 128) Bt[buf][bn16 + 16][bk] = rb1[7];
            if (bn16 == 0)       Bt[buf][0][bk]         = (bf16)0.f;
        } else {                                       // kx==2, shift left: n = w - 1
            #pragma unroll
            for (int j = 0; j < 7; ++j) Bt[buf][bn16 + j][bk] = rb0[j + 1];
            #pragma unroll
            for (int j = 0; j < 8; ++j) Bt[buf][bn16 + 7 + j][bk] = rb1[j];
            if (bn16 > 0)         Bt[buf][bn16 - 1][bk] = rb0[0];
            if (bn16 == 112)      Bt[buf][127][bk]      = (bf16)0.f;
        }
    };

    {   // prologue: fill buffer 0
        bf16x8 a0, a1, b0, b1; bool rk;
        load_stage(0, a0, a1, b0, b1, rk);
        store_stage(0, 0, a0, a1, b0, b1, rk);
    }
    __syncthreads();

    for (int iter = 0; iter < 144; ++iter) {
        int buf = iter & 1;
        bool more = (iter + 1) < 144;
        bf16x8 na0, na1, nb0, nb1; bool nrk = false;
        if (more) load_stage(iter + 1, na0, na1, nb0, nb1, nrk);

        Frag a[2], b[4];
        #pragma unroll
        for (int f = 0; f < 2; ++f) {
            int row = mbase + f * 16 + ml;
            a[f].h[0] = *(const bf16x8*)&At[buf][row][sel * 8];
            a[f].h[1] = *(const bf16x8*)&At[buf][row][16 + sel * 8];
        }
        #pragma unroll
        for (int f = 0; f < 4; ++f) {
            int col = nbase + f * 16 + ml;
            b[f].h[0] = *(const bf16x8*)&Bt[buf][col][sel * 8];
            b[f].h[1] = *(const bf16x8*)&Bt[buf][col][16 + sel * 8];
        }
        #pragma unroll
        for (int fm = 0; fm < 2; ++fm)
            #pragma unroll
            for (int fn = 0; fn < 4; ++fn)
                acc[fm][fn] = __builtin_amdgcn_wmma_f32_16x16x32_bf16(
                    false, a[fm].v, false, b[fn].v, (short)0, acc[fm][fn],
                    false, false);

        if (more) store_stage(buf ^ 1, iter + 1, na0, na1, nb0, nb1, nrk);
        __syncthreads();
    }

    // epilogue: bias + relu -> bf16 h in NCHW
    #pragma unroll
    for (int fm = 0; fm < 2; ++fm)
        #pragma unroll
        for (int fn = 0; fn < 4; ++fn)
            #pragma unroll
            for (int r = 0; r < 8; ++r) {
                int co = co0 + mbase + fm * 16 + sel * 8 + r;
                int p  = p0 + nbase + fn * 16 + ml;
                float v = acc[fm][fn][r] + conv_b[co];
                hbuf[(size_t)co * NPIX + p] = (bf16)fmaxf(v, 0.f);
            }
}

// ------------------------------------- fused 1x1 heads: out[64][16384] = wc x h
__global__ __launch_bounds__(256) void k_gemm2(const bf16* __restrict__ hbuf,
                                               const bf16* __restrict__ wc,
                                               const float* __restrict__ bbox_b,
                                               const float* __restrict__ score_b,
                                               float* __restrict__ bboxf,
                                               float* __restrict__ scoref) {
    __shared__ __attribute__((aligned(16))) bf16 At[64][40];
    __shared__ __attribute__((aligned(16))) bf16 Bt[128][40];

    int tid = threadIdx.x;
    int n0  = blockIdx.x * 128;
    int wv = tid >> 5, lane = tid & 31;
    int wm = wv & 1, wn = wv >> 1;
    int mbase = wm * 32, nbase = wn * 32;
    int ml = lane & 15, sel = lane >> 4;

    v8f acc[2][2] = {};
    int am = tid >> 2, ah = (tid & 3) * 8;
    int bk = tid >> 3, bn16 = (tid & 7) * 16;

    for (int ci0 = 0; ci0 < CIN; ci0 += 32) {
        __syncthreads();
        *(bf16x8*)&At[am][ah] = *(const bf16x8*)(wc + am * CIN + ci0 + ah);
        {
            const bf16* src = hbuf + (size_t)(ci0 + bk) * NPIX + n0 + bn16;
            bf16x8 v0 = *(const bf16x8*)src;
            bf16x8 v1 = *(const bf16x8*)(src + 8);
            #pragma unroll
            for (int j = 0; j < 8; ++j) {
                Bt[bn16 + j][bk]     = v0[j];
                Bt[bn16 + 8 + j][bk] = v1[j];
            }
        }
        __syncthreads();

        Frag a[2], b[2];
        #pragma unroll
        for (int f = 0; f < 2; ++f) {
            int row = mbase + f * 16 + ml;
            a[f].h[0] = *(const bf16x8*)&At[row][sel * 8];
            a[f].h[1] = *(const bf16x8*)&At[row][16 + sel * 8];
        }
        #pragma unroll
        for (int f = 0; f < 2; ++f) {
            int col = nbase + f * 16 + ml;
            b[f].h[0] = *(const bf16x8*)&Bt[col][sel * 8];
            b[f].h[1] = *(const bf16x8*)&Bt[col][16 + sel * 8];
        }
        #pragma unroll
        for (int fm = 0; fm < 2; ++fm)
            #pragma unroll
            for (int fn = 0; fn < 2; ++fn)
                acc[fm][fn] = __builtin_amdgcn_wmma_f32_16x16x32_bf16(
                    false, a[fm].v, false, b[fn].v, (short)0, acc[fm][fn],
                    false, false);
    }

    #pragma unroll
    for (int fm = 0; fm < 2; ++fm)
        #pragma unroll
        for (int fn = 0; fn < 2; ++fn)
            #pragma unroll
            for (int r = 0; r < 8; ++r) {
                int m = mbase + fm * 16 + sel * 8 + r;
                int n = n0 + nbase + fn * 16 + ml;
                float v = acc[fm][fn][r];
                if (m < 36)      bboxf[m * NPIX + n]         = fmaxf(v + bbox_b[m], 0.f);
                else if (m < 54) scoref[(m - 36) * NPIX + n] = fmaxf(v + score_b[m - 36], 0.f);
            }
}

// ---------------------------------------------------------------- anchors/IoU
__device__ __forceinline__ void anchor_box(int a, float stepx, float stepy,
                                           float& x1, float& y1, float& x2, float& y2) {
    int r = a / 1152;
    int c = (a / 9) & 127;
    int k = a % 9;
    x1 = r * stepx;
    y1 = c * stepy;
    float side = (float)(128 << (k / 3));
    int rm = k % 3;
    float w, h;
    if (rm == 0)      { w = side;                 h = side; }
    else if (rm == 1) { w = side * 1.41421356f;   h = side * 0.70710678f; }
    else              { w = side * 0.70710678f;   h = side * 1.41421356f; }
    x2 = x1 + w; y2 = y1 + h;
}

__global__ void k_iou(const float* __restrict__ target, const int* __restrict__ imw,
                      const int* __restrict__ imh, int* __restrict__ tidx,
                      float* __restrict__ ps, float* __restrict__ ns) {
    __shared__ float4 tg[NT];
    int tid = threadIdx.x;
    if (tid < NT) tg[tid] = ((const float4*)target)[tid];
    __syncthreads();
    int a = blockIdx.x * blockDim.x + tid;
    if (a >= NANCH) return;
    float stepx = (float)imw[0] / 127.f, stepy = (float)imh[0] / 127.f;
    float x1, y1, x2, y2;
    anchor_box(a, stepx, stepy, x1, y1, x2, y2);
    float aa = (x2 - x1) * (y2 - y1);
    float best = -1.f; int bi = 0;
    #pragma unroll 4
    for (int t = 0; t < NT; ++t) {
        float4 tb = tg[t];
        float iw = fminf(tb.z, x2) - fmaxf(tb.x, x1);
        float ih = fminf(tb.w, y2) - fmaxf(tb.y, y1);
        iw = fmaxf(iw, 0.f); ih = fmaxf(ih, 0.f);
        float inter = iw * ih;
        float at = (tb.z - tb.x) * (tb.w - tb.y);
        float iou = inter / (at + aa - inter);
        if (iou > best) { best = iou; bi = t; }
    }
    tidx[a] = bi;
    ps[a] = (best > 0.7f) ? best : -1.f;
    ns[a] = (best < 0.3f) ? 1.f - best : -1.f;
}

// ------------------------------------------------- top-k (single workgroup)
__global__ __launch_bounds__(1024) void k_select(float* __restrict__ ps,
                                                 float* __restrict__ ns,
                                                 int* __restrict__ pidx,
                                                 float* __restrict__ pval,
                                                 int* __restrict__ nidx,
                                                 float* __restrict__ nval,
                                                 int* __restrict__ npos) {
    __shared__ float sv[1024];
    __shared__ int   si[1024];
    int t = threadIdx.x;

    for (int it = 0; it < 128; ++it) {                 // top-128 positives
        float bv = -3e30f; int bi = 0x7fffffff;
        for (int i = t; i < NANCH; i += 1024) {
            float v = ps[i];
            if (v > bv) { bv = v; bi = i; }
        }
        sv[t] = bv; si[t] = bi; __syncthreads();
        for (int s = 512; s > 0; s >>= 1) {
            if (t < s && (sv[t + s] > sv[t] ||
                          (sv[t + s] == sv[t] && si[t + s] < si[t]))) {
                sv[t] = sv[t + s]; si[t] = si[t + s];
            }
            __syncthreads();
        }
        if (t == 0) { pidx[it] = si[0]; pval[it] = sv[0]; ps[si[0]] = -4e30f; }
        __syncthreads();
    }
    for (int it = 0; it < 256; ++it) {                 // top-256 negatives
        float bv = -3e30f; int bi = 0x7fffffff;
        for (int i = t; i < NANCH; i += 1024) {
            float v = ns[i];
            if (v > bv) { bv = v; bi = i; }
        }
        sv[t] = bv; si[t] = bi; __syncthreads();
        for (int s = 512; s > 0; s >>= 1) {
            if (t < s && (sv[t + s] > sv[t] ||
                          (sv[t + s] == sv[t] && si[t + s] < si[t]))) {
                sv[t] = sv[t + s]; si[t] = si[t + s];
            }
            __syncthreads();
        }
        if (t == 0) { nidx[it] = si[0]; nval[it] = sv[0]; ns[si[0]] = -4e30f; }
        __syncthreads();
    }
    si[t] = (t < 128 && pval[t] > 0.f) ? 1 : 0;
    __syncthreads();
    for (int s = 512; s > 0; s >>= 1) {
        if (t < s) si[t] += si[t + s];
        __syncthreads();
    }
    if (t == 0) npos[0] = si[0];
}

// ---------------------------------------------------------------- final loss
__global__ __launch_bounds__(512) void k_loss(const float* __restrict__ scoref,
                                              const float* __restrict__ bboxf,
                                              const float* __restrict__ target,
                                              const int* __restrict__ tidx,
                                              const int* __restrict__ pidx,
                                              const float* __restrict__ pval,
                                              const int* __restrict__ nidx,
                                              const float* __restrict__ nval,
                                              const int* __restrict__ npos,
                                              const int* __restrict__ imw,
                                              const int* __restrict__ imh,
                                              float* __restrict__ out) {
    __shared__ float r0[512];
    __shared__ float r1[512];
    int t = threadIdx.x;
    int np = npos[0];

    float ce = 0.f, va = 0.f;
    if (t < 384) {
        int idx; float valid; bool isp = (t < 128);
        if (isp) { idx = pidx[t]; valid = (pval[t] > 0.f) ? 1.f : 0.f; }
        else {
            int j = t - 128; idx = nidx[j];
            valid = (nval[j] > 0.f && j < 256 - np) ? 1.f : 0.f;
        }
        int c = idx >> 13;                 // score row -> channel, pixel pair
        int p = (idx & 8191) * 2;
        float s0 = scoref[c * NPIX + p];
        float s1 = scoref[c * NPIX + p + 1];
        float mx = fmaxf(s0, s1);
        float lse = mx + logf(expf(s0 - mx) + expf(s1 - mx));
        float ls = (isp ? s0 : s1) - lse;
        ce = -ls * valid;
        va = valid;
    }
    r0[t] = ce; r1[t] = va; __syncthreads();
    for (int s = 256; s > 0; s >>= 1) {
        if (t < s) { r0[t] += r0[t + s]; r1[t] += r1[t + s]; }
        __syncthreads();
    }
    float CE = 0.f, V = 0.f;
    if (t == 0) { CE = r0[0]; V = r1[0]; }
    __syncthreads();

    float sl = 0.f, pm = 0.f;
    if (t < 128 && pval[t] > 0.f) {
        int a = pidx[t];
        float stepx = (float)imw[0] / 127.f, stepy = (float)imh[0] / 127.f;
        float x1, y1, x2, y2;
        anchor_box(a, stepx, stepy, x1, y1, x2, y2);
        float4 tb = ((const float4*)target)[tidx[a]];
        float aw = x2 - x1, ahh = y2 - y1;
        float acx = x1 + aw * 0.5f, acy = y1 + ahh * 0.5f;
        float bw = tb.z - tb.x, bh = tb.w - tb.y;
        float bcx = tb.x + bw * 0.5f, bcy = tb.y + bh * 0.5f;
        float tr[4] = { (bcx - acx) / aw, (bcy - acy) / ahh,
                        logf(bw / aw),    logf(bh / ahh) };
        int c = a >> 12;                   // bbox row -> channel, 4-pixel group
        int p = (a & 4095) * 4;
        #pragma unroll
        for (int j = 0; j < 4; ++j) {
            float d = bboxf[c * NPIX + p + j] - tr[j];
            float ad = fabsf(d);
            sl += (ad < 1.f) ? 0.5f * d * d : ad - 0.5f;
        }
        pm = 1.f;
    }
    r0[t] = sl; r1[t] = pm; __syncthreads();
    for (int s = 256; s > 0; s >>= 1) {
        if (t < s) { r0[t] += r0[t + s]; r1[t] += r1[t + s]; }
        __syncthreads();
    }
    if (t == 0) {
        float score_loss = CE / fmaxf(V, 1.f);
        float reg_loss   = r0[0] / fmaxf(r1[0] * 4.f, 1.f);
        out[0] = score_loss + 10.f * reg_loss;
    }
}

// ---------------------------------------------------------------- launcher
extern "C" void kernel_launch(void* const* d_in, const int* in_sizes, int n_in,
                              void* d_out, int out_size, void* d_ws, size_t ws_size,
                              hipStream_t stream) {
    const float* x       = (const float*)d_in[0];
    const float* target  = (const float*)d_in[1];
    const float* conv_w  = (const float*)d_in[2];
    const float* conv_b  = (const float*)d_in[3];
    const float* bbox_w  = (const float*)d_in[4];
    const float* bbox_b  = (const float*)d_in[5];
    const float* score_w = (const float*)d_in[6];
    const float* score_b = (const float*)d_in[7];
    const int*   img_h   = (const int*)d_in[8];
    const int*   img_w   = (const int*)d_in[9];

    char* ws = (char*)d_ws;
    bf16*  xb     = (bf16*) (ws);                    // 16 MB
    bf16*  wbf    = (bf16*) (ws + 16777216);         // 4.5 MB
    bf16*  wc     = (bf16*) (ws + 21495808);         // 64 KB
    bf16*  hbuf   = (bf16*) (ws + 21561344);         // 16 MB
    float* bboxf  = (float*)(ws + 38338560);         // 2.25 MB
    float* scoref = (float*)(ws + 40697856);         // 1.125 MB
    int*   tidx   = (int*)  (ws + 41877504);         // 576 KB
    float* ps     = (float*)(ws + 42467328);         // 576 KB
    float* ns     = (float*)(ws + 43057152);         // 576 KB
    int*   pidx   = (int*)  (ws + 43646976);
    float* pval   = (float*)(ws + 43647488);
    int*   nidx   = (int*)  (ws + 43648000);
    float* nval   = (float*)(ws + 43649024);
    int*   npos   = (int*)  (ws + 43650048);

    k_cvt_x<<<(CIN * NPIX / 4 + 255) / 256, 256, 0, stream>>>(x, xb, CIN * NPIX);
    k_cvt_w<<<(CIN * 9 * CIN + 255) / 256, 256, 0, stream>>>(conv_w, bbox_w, score_w,
                                                             wbf, wc);
    k_conv<<<512, 256, 0, stream>>>(xb, wbf, conv_b, hbuf);
    k_gemm2<<<NPIX / 128, 256, 0, stream>>>(hbuf, wc, bbox_b, score_b, bboxf, scoref);
    k_iou<<<(NANCH + 255) / 256, 256, 0, stream>>>(target, img_w, img_h, tidx, ps, ns);
    k_select<<<1, 1024, 0, stream>>>(ps, ns, pidx, pval, nidx, nval, npos);
    k_loss<<<1, 512, 0, stream>>>(scoref, bboxf, target, tidx, pidx, pval,
                                  nidx, nval, npos, img_w, img_h, (float*)d_out);
}
